// AttentionAggregate_87428354277737
// MI455X (gfx1250) — matrix operand order repaired and checked by reading, not compiled, not run on hardware
//
#include <hip/hip_runtime.h>
#include <hip/hip_bf16.h>

// ---------------------------------------------------------------------------
// Problem constants (match reference)
// ---------------------------------------------------------------------------
constexpr int BATCH = 2;
constexpr int SEQ   = 8192;   // N
constexpr int NCL   = 512;    // NC (clusters)
constexpr int DIM   = 512;    // D
constexpr int NH    = 16;     // H
constexpr int HDIM  = 32;     // D / H
constexpr float SCALE = 0.17677669529663687f;   // 32^-0.5
constexpr float EPS   = 1e-6f;

constexpr int KCHUNKS = 16;                 // key chunks per (b,h)
constexpr int KEYS_PER_WG = SEQ / KCHUNKS;  // 512
constexpr int KB = 32;                      // keys per inner step

// ---------------------------------------------------------------------------
// WMMA / TDM types
// ---------------------------------------------------------------------------
typedef __attribute__((ext_vector_type(16))) __bf16 v16bf;
typedef __attribute__((ext_vector_type(8)))  float  v8f;
typedef __attribute__((ext_vector_type(4)))  unsigned tdm_g0_t;
typedef __attribute__((ext_vector_type(8)))  int      tdm_g1_t;
typedef __attribute__((ext_vector_type(4)))  int      tdm_g2_t;

union FragU {                 // one WMMA bf16 operand = 8 dwords (16 bf16)
  unsigned u[8];
  v16bf v;
};

static __device__ __forceinline__ unsigned short f2bf(float f) {
  unsigned u = __float_as_uint(f);
  u += 0x7FFFu + ((u >> 16) & 1u);       // round-to-nearest-even
  return (unsigned short)(u >> 16);
}
static __device__ __forceinline__ unsigned pack2(float a, float b) {
  return (unsigned)f2bf(a) | ((unsigned)f2bf(b) << 16);
}
static __device__ __forceinline__ float bf2f(unsigned short s) {
  return __uint_as_float(((unsigned)s) << 16);
}

// ---------------------------------------------------------------------------
// TDM: 2D tile load, 2-byte elements, LDS padding of 1 dword per 16 dwords
// (=> 34-ushort padded row stride, matching the LDS layouts below).
// tensor_d0/stride0/tile dims in elements.
// 6-arg toolchain form: (g0, g1, g2, g3, g_extra, cpol).
// ---------------------------------------------------------------------------
static __device__ __forceinline__ void tdm_load_2d_bf16(
    const void* gptr, unsigned lds_byte_off,
    unsigned tensor_d0, unsigned tensor_d1, unsigned stride0_elems,
    unsigned tile_d0, unsigned tile_d1)
{
  unsigned long long ga = (unsigned long long)gptr;
  tdm_g0_t g0;
  g0[0] = 1u;                                              // count=1 (valid)
  g0[1] = lds_byte_off;                                    // lds_addr
  g0[2] = (unsigned)ga;                                    // global_addr lo
  g0[3] = (unsigned)((ga >> 32) & 0x01FFFFFFull) | (2u << 30);  // hi | type=2
  tdm_g1_t g1;
  g1[0] = (int)((1u << 16) | (1u << 20) | (3u << 22));     // 2B, pad 1dw/16dw
  g1[1] = (int)((tensor_d0 & 0xFFFFu) << 16);              // tensor_dim0 lo16
  g1[2] = (int)((tensor_d0 >> 16) | ((tensor_d1 & 0xFFFFu) << 16));
  g1[3] = (int)((tensor_d1 >> 16) | (tile_d0 << 16));      // tile_dim0
  g1[4] = (int)(tile_d1 & 0xFFFFu);                        // tile_dim1 (dim2=0)
  g1[5] = (int)stride0_elems;                              // dim0_stride lo32
  g1[6] = 0;
  g1[7] = 0;
  tdm_g2_t gz4 = {0, 0, 0, 0};
  tdm_g1_t gz8 = {0, 0, 0, 0, 0, 0, 0, 0};
  __builtin_amdgcn_tensor_load_to_lds(g0, g1, gz4, gz4, gz8, 0);
}

// ---------------------------------------------------------------------------
// Generic GEMM: C[M,N] = A[M,K](f32) @ W[K,N](f32), bf16 WMMA, f32 acc.
// 128x128x32 tiles, 256 threads = 8 waves, each wave owns a 32x64 sub-tile.
// LDS tiles are stored pair-packed (uint = {bf16 k, bf16 k+1}) so every
// fragment dword is a single aligned ds_load_b32.
// ---------------------------------------------------------------------------
template <int OUT_BF16>
__global__ __launch_bounds__(256) void gemm_wmma_bf16(
    const float* __restrict__ A, const float* __restrict__ W,
    void* __restrict__ C, int M, int K, int Nw)
{
  constexpr int BM = 128, BN = 128, BKt = 32;
  constexpr int LDA = 17;    // uints per A row (16 pairs + 1 pad)
  constexpr int LDB = 129;   // uints per packed-B row (128 + 1 pad)
  __shared__ unsigned Atp[BM * LDA];          // [row][kpair]
  __shared__ unsigned Btp[(BKt / 2) * LDB];   // [kpair][col]

  const int tid  = threadIdx.x;
  const int lane = tid & 31;
  const int wave = tid >> 5;
  const int wr   = wave & 3;   // rows wr*32
  const int wc   = wave >> 2;  // cols wc*64
  const int hi   = (lane >> 4) & 1;
  const int l15  = lane & 15;
  const int bm = blockIdx.y, bn = blockIdx.x;

  v8f acc[2][4];
#pragma unroll
  for (int rt = 0; rt < 2; ++rt)
#pragma unroll
    for (int ct = 0; ct < 4; ++ct) { v8f z = {}; acc[rt][ct] = z; }

  for (int k0 = 0; k0 < K; k0 += BKt) {
    // A tile: 128x32 f32 -> pair-packed bf16, float4 global loads
#pragma unroll
    for (int i = 0; i < 4; ++i) {
      int idx = tid + i * 256;              // 1024 chunks of 4 floats
      int r = idx >> 3, c4 = (idx & 7) * 4;
      const float4 f = *(const float4*)&A[(size_t)(bm * BM + r) * K + k0 + c4];
      Atp[r * LDA + (c4 >> 1)]     = pack2(f.x, f.y);
      Atp[r * LDA + (c4 >> 1) + 1] = pack2(f.z, f.w);
    }
    // B tile: 32x128 f32 -> pair-packed over K
#pragma unroll
    for (int i = 0; i < 8; ++i) {
      int idx = tid + i * 256;              // 2048 packed elements
      int kp = idx >> 7, n = idx & 127;
      size_t g = (size_t)(k0 + 2 * kp) * Nw + bn * BN + n;
      Btp[kp * LDB + n] = pack2(W[g], W[g + Nw]);
    }
    __syncthreads();

    FragU af[2], bfr[4];
#pragma unroll
    for (int rt = 0; rt < 2; ++rt) {
      int m = wr * 32 + rt * 16 + l15;
#pragma unroll
      for (int i = 0; i < 8; ++i)
        af[rt].u[i] = Atp[m * LDA + i + hi * 8];
    }
#pragma unroll
    for (int ct = 0; ct < 4; ++ct) {
      int n = wc * 64 + ct * 16 + l15;
#pragma unroll
      for (int i = 0; i < 8; ++i)
        bfr[ct].u[i] = Btp[(i + hi * 8) * LDB + n];
    }
#pragma unroll
    for (int rt = 0; rt < 2; ++rt)
#pragma unroll
      for (int ct = 0; ct < 4; ++ct)
        acc[rt][ct] = __builtin_amdgcn_wmma_f32_16x16x32_bf16(
            false, af[rt].v, false, bfr[ct].v, (short)0, acc[rt][ct],
            false, false);
    __syncthreads();
  }

#pragma unroll
  for (int rt = 0; rt < 2; ++rt)
#pragma unroll
    for (int ct = 0; ct < 4; ++ct) {
      int n = bn * BN + wc * 64 + ct * 16 + l15;
#pragma unroll
      for (int r = 0; r < 8; ++r) {
        int m = bm * BM + wr * 32 + rt * 16 + r + hi * 8;
        if (OUT_BF16)
          ((unsigned short*)C)[(size_t)m * Nw + n] = f2bf(acc[rt][ct][r]);
        else
          ((float*)C)[(size_t)m * Nw + n] = acc[rt][ct][r];
      }
    }
}

// ---------------------------------------------------------------------------
// Fused cluster attention. One WG per (b, h, 512-key chunk). 256 thr, 8 waves.
// q and k tiles staged by the Tensor Data Mover; v staged pair-packed by VALU.
// Per 32-key step: S = q @ k^T (WMMA), column softmax over the 512-cluster
// axis, U += P @ v (WMMA, persistent acc), ts += row sums of P.
// ---------------------------------------------------------------------------
constexpr int SLDF = 33;  // float row stride for S
constexpr int SLDH = 34;  // ushort row stride for q/P/k (17 dwords; TDM pad)
constexpr size_t ATTN_SMEM =
    (size_t)(NCL * SLDF + 8 * 32 + 32 + 32) * 4 +   // floats: S, red, colm/s
    (size_t)(KB / 2) * 33 * 4 +                     // vbp (uint, pair-packed)
    (size_t)(NCL * SLDH * 2 + KB * SLDH) * 2;       // qs, Pl, kb (ushort)

__global__ __launch_bounds__(256) void attn_cluster(
    const unsigned short* __restrict__ qbf,   // [BATCH*NCL, DIM] bf16
    const unsigned short* __restrict__ kvbf,  // [BATCH*SEQ, 2*DIM] bf16
    float* __restrict__ U,                    // [BATCH,NH,NCL,HDIM]
    float* __restrict__ ts)                   // [BATCH,NH,NCL]
{
  const int bh    = blockIdx.x / KCHUNKS;
  const int chunk = blockIdx.x % KCHUNKS;
  const int b = bh / NH, h = bh % NH;

  extern __shared__ char smem[];
  float*          Sl   = (float*)smem;                   // NCL x SLDF
  float*          red  = Sl + NCL * SLDF;                // 8 x 32
  float*          colm = red + 8 * 32;                   // 32
  float*          cols = colm + 32;                      // 32
  unsigned*       vbp  = (unsigned*)(cols + 32);         // 16 x 33 (pairs)
  unsigned short* qs   = (unsigned short*)(vbp + 16 * 33); // NCL x SLDH
  unsigned short* Pl   = qs + NCL * SLDH;                // NCL x SLDH
  unsigned short* kb   = Pl + NCL * SLDH;                // KB x SLDH
  const unsigned* qs32 = (const unsigned*)qs;
  const unsigned* Pl32 = (const unsigned*)Pl;
  const unsigned* kb32 = (const unsigned*)kb;

  const int tid  = threadIdx.x;
  const int lane = tid & 31;
  const int wave = tid >> 5;
  const int hi   = (lane >> 4) & 1;
  const int l15  = lane & 15;

  // TDM: stage q[b,h] (512 x 32 bf16, row stride DIM) into padded LDS
  if (wave == 0) {
    tdm_load_2d_bf16(qbf + (size_t)(b * NCL) * DIM + h * HDIM,
                     (unsigned)(unsigned long long)(void*)qs,
                     HDIM, NCL, DIM, HDIM, NCL);
  }

  v8f uacc[4][2];
#pragma unroll
  for (int rt = 0; rt < 4; ++rt)
#pragma unroll
    for (int ct = 0; ct < 2; ++ct) { v8f z = {}; uacc[rt][ct] = z; }
  float tsa0 = 0.f, tsa1 = 0.f;          // token-size partials, rows 2*tid(+1)

  for (int s = 0; s < KEYS_PER_WG / KB; ++s) {
    const int kk0 = chunk * KEYS_PER_WG + s * KB;
    // TDM: stage k block (32 keys x 32 dims, row stride 2*DIM)
    if (wave == 0) {
      tdm_load_2d_bf16(kvbf + (size_t)(b * SEQ + kk0) * (2 * DIM) + h * HDIM,
                       (unsigned)(unsigned long long)(void*)kb,
                       HDIM, KB, 2 * DIM, HDIM, KB);
    }
    // VALU: stage v block pair-packed over keys: vbp[kp][d] = {v[2kp,d], v[2kp+1,d]}
    for (int i = tid; i < (KB / 2) * HDIM; i += 256) {
      int kp = i >> 5, c = i & 31;
      size_t base = (size_t)(b * SEQ + kk0 + 2 * kp) * (2 * DIM) + DIM + h * HDIM + c;
      vbp[kp * 33 + c] =
          (unsigned)kvbf[base] | ((unsigned)kvbf[base + 2 * DIM] << 16);
    }
    if (wave == 0) __builtin_amdgcn_s_wait_tensorcnt(0);
    __syncthreads();

    // ---- S = scale * q @ k^T  (512 x 32), 8 WMMAs per wave ----
#pragma unroll
    for (int rt = 0; rt < 4; ++rt) {
      const int m0 = (wave * 4 + rt) * 16;
      FragU af;
#pragma unroll
      for (int i = 0; i < 8; ++i)
        af.u[i] = qs32[(m0 + l15) * 17 + i + hi * 8];
#pragma unroll
      for (int ct = 0; ct < 2; ++ct) {
        FragU bfr;                     // B[d][key] = kb[key][d], pairs in-row
        int n = ct * 16 + l15;
#pragma unroll
        for (int i = 0; i < 8; ++i)
          bfr.u[i] = kb32[n * 17 + i + hi * 8];
        v8f sa = {};
        sa = __builtin_amdgcn_wmma_f32_16x16x32_bf16(
            false, af.v, false, bfr.v, (short)0, sa, false, false);
#pragma unroll
        for (int r = 0; r < 8; ++r)
          Sl[(m0 + r + hi * 8) * SLDF + n] = sa[r] * SCALE;
      }
    }
    __syncthreads();

    // ---- column softmax over the 512-cluster axis ----
    {
      const int c = tid & 31, seg = tid >> 5;      // 8 segs x 64 rows
      float mx = -3.4e38f;
      for (int r = 0; r < 64; ++r)
        mx = fmaxf(mx, Sl[(seg * 64 + r) * SLDF + c]);
      red[seg * 32 + c] = mx;
      __syncthreads();
      if (tid < 32) {
        float m2 = red[tid];
        for (int g = 1; g < 8; ++g) m2 = fmaxf(m2, red[g * 32 + tid]);
        colm[tid] = m2;
      }
      __syncthreads();
      const float cm = colm[c];
      float sum = 0.f;
      for (int r = 0; r < 64; ++r) {
        int idx = (seg * 64 + r) * SLDF + c;
        float e = __expf(Sl[idx] - cm);
        Sl[idx] = e;
        sum += e;
      }
      red[seg * 32 + c] = sum;
      __syncthreads();
      if (tid < 32) {
        float s2 = red[tid];
        for (int g = 1; g < 8; ++g) s2 += red[g * 32 + tid];
        cols[tid] = s2;
      }
      __syncthreads();
      const float inv = 1.0f / cols[c];
      for (int r = 0; r < 64; ++r)
        Pl[(seg * 64 + r) * SLDH + c] =
            f2bf(Sl[(seg * 64 + r) * SLDF + c] * inv);
    }
    __syncthreads();

    // ---- token sizes: row sums of P (2 rows / thread) ----
    {
      const int r0 = tid * 2;
      float a = 0.f, b2 = 0.f;
      for (int c = 0; c < KB; ++c) {
        a  += bf2f(Pl[r0 * SLDH + c]);
        b2 += bf2f(Pl[(r0 + 1) * SLDH + c]);
      }
      tsa0 += a; tsa1 += b2;
    }

    // ---- U += P @ v  (512 x 32), 8 WMMAs per wave, persistent acc ----
#pragma unroll
    for (int rt = 0; rt < 4; ++rt) {
      const int m0 = (wave * 4 + rt) * 16;
      FragU af;
#pragma unroll
      for (int i = 0; i < 8; ++i)
        af.u[i] = Pl32[(m0 + l15) * 17 + i + hi * 8];
#pragma unroll
      for (int ct = 0; ct < 2; ++ct) {
        FragU bfr;                     // B[key][d], key-pairs pre-packed
        int n = ct * 16 + l15;
#pragma unroll
        for (int i = 0; i < 8; ++i)
          bfr.u[i] = vbp[(i + hi * 8) * 33 + n];
        uacc[rt][ct] = __builtin_amdgcn_wmma_f32_16x16x32_bf16(
            false, af.v, false, bfr.v, (short)0, uacc[rt][ct], false, false);
      }
    }
    __syncthreads();
  }

  // ---- merge partials ----
#pragma unroll
  for (int rt = 0; rt < 4; ++rt)
#pragma unroll
    for (int ct = 0; ct < 2; ++ct) {
      int d = ct * 16 + l15;
#pragma unroll
      for (int r = 0; r < 8; ++r) {
        int m = (wave * 4 + rt) * 16 + r + hi * 8;
        atomicAdd(&U[((size_t)(b * NH + h) * NCL + m) * HDIM + d],
                  uacc[rt][ct][r]);
      }
    }
  atomicAdd(&ts[(size_t)(b * NH + h) * NCL + tid * 2],     tsa0);
  atomicAdd(&ts[(size_t)(b * NH + h) * NCL + tid * 2 + 1], tsa1);
}

// ---------------------------------------------------------------------------
// Small helper kernels
// ---------------------------------------------------------------------------
__global__ __launch_bounds__(256) void zero_f32(float* p, int n) {
  int i = blockIdx.x * 256 + threadIdx.x;
  if (i < n) p[i] = 0.f;
}

__global__ __launch_bounds__(256) void normalize_k(
    const float* __restrict__ U, const float* __restrict__ ts,
    float* __restrict__ Ub)
{
  int i = blockIdx.x * 256 + threadIdx.x;           // over BATCH*NCL*DIM
  if (i >= BATCH * NCL * DIM) return;
  int d = i % DIM;
  int q = (i / DIM) % NCL;
  int b = i / (DIM * NCL);
  int h = d / HDIM, dd = d % HDIM;
  float t = ts[(size_t)(b * NH + h) * NCL + q];
  Ub[i] = U[((size_t)(b * NH + h) * NCL + q) * HDIM + dd] / (t + EPS);
}

__global__ __launch_bounds__(256) void tsmean_k(
    const float* __restrict__ ts, float* __restrict__ outTs)
{
  int i = blockIdx.x * 256 + threadIdx.x;           // over BATCH*NCL
  if (i >= BATCH * NCL) return;
  int b = i / NCL, q = i % NCL;
  float s = 0.f;
  for (int h = 0; h < NH; ++h) s += ts[(size_t)(b * NH + h) * NCL + q];
  outTs[i] = s * (1.0f / NH);
}

// ---------------------------------------------------------------------------
// Host launcher
// ---------------------------------------------------------------------------
extern "C" void kernel_launch(void* const* d_in, const int* in_sizes, int n_in,
                              void* d_out, int out_size, void* d_ws, size_t ws_size,
                              hipStream_t stream) {
  const float* x        = (const float*)d_in[0];   // [B,N,D]
  const float* clusters = (const float*)d_in[1];   // [B,NC,D]
  const float* Wq       = (const float*)d_in[2];   // [D,D]
  const float* Wkv      = (const float*)d_in[3];   // [D,2D]
  const float* Wout     = (const float*)d_in[4];   // [D,D]
  float* out = (float*)d_out;                      // [B*NC*D] + [B*NC]

  // workspace carve
  float* U  = (float*)d_ws;                        // B*H*NC*HD
  float* ts = U + (size_t)BATCH * NH * NCL * HDIM; // B*H*NC
  float* Ub = ts + (size_t)BATCH * NH * NCL;       // B*NC*D  (f32)
  unsigned short* qbf  = (unsigned short*)(Ub + (size_t)BATCH * NCL * DIM);
  unsigned short* kvbf = qbf + (size_t)BATCH * NCL * DIM;   // B*N*2D bf16

  // 0) zero accumulators (U and ts are contiguous)
  int nz = BATCH * NH * NCL * HDIM + BATCH * NH * NCL;
  zero_f32<<<(nz + 255) / 256, 256, 0, stream>>>(U, nz);

  // 1) q = clusters @ Wq        -> bf16 [B*NC, D]
  {
    dim3 g(DIM / 128, (BATCH * NCL) / 128);
    gemm_wmma_bf16<1><<<g, 256, 0, stream>>>(clusters, Wq, qbf,
                                             BATCH * NCL, DIM, DIM);
  }
  // 2) kv = x @ Wkv             -> bf16 [B*N, 2D]
  {
    dim3 g((2 * DIM) / 128, (BATCH * SEQ) / 128);
    gemm_wmma_bf16<1><<<g, 256, 0, stream>>>(x, Wkv, kvbf,
                                             BATCH * SEQ, DIM, 2 * DIM);
  }
  // 3) fused attention: U, ts
  attn_cluster<<<BATCH * NH * KCHUNKS, 256, ATTN_SMEM, stream>>>(
      qbf, kvbf, U, ts);

  // 4) normalize + head-merge to Ub (f32), token-size mean to output tail
  {
    int nn = BATCH * NCL * DIM;
    normalize_k<<<(nn + 255) / 256, 256, 0, stream>>>(U, ts, Ub);
    tsmean_k<<<(BATCH * NCL + 255) / 256, 256, 0, stream>>>(
        ts, out + (size_t)BATCH * NCL * DIM);
  }
  // 5) out = Ub @ Wout          -> f32 [B*NC, D]
  {
    dim3 g(DIM / 128, (BATCH * NCL) / 128);
    gemm_wmma_bf16<0><<<g, 256, 0, stream>>>(Ub, Wout, out,
                                             BATCH * NCL, DIM, DIM);
  }
}
